// NicheAttentionNetwork_26594437497340
// MI455X (gfx1250) — matrix operand order repaired and checked by reading, not compiled, not run on hardware
//
#include <hip/hip_runtime.h>

typedef __attribute__((ext_vector_type(16))) __bf16 v16bf;
typedef __attribute__((ext_vector_type(8)))  __bf16 v8bf;
typedef __attribute__((ext_vector_type(8)))  float  v8f;
typedef __attribute__((ext_vector_type(4)))  unsigned short ushort4v;

// ---------- helpers ----------
__device__ __forceinline__ __bf16 f2bf(float f) {
    unsigned u = __builtin_bit_cast(unsigned, f);
    unsigned r = (u + 0x7FFFu + ((u >> 16) & 1u)) >> 16;
    unsigned short s = (unsigned short)r;
    return __builtin_bit_cast(__bf16, s);
}
__device__ __forceinline__ float bf2f(__bf16 h) {
    unsigned u = ((unsigned)__builtin_bit_cast(unsigned short, h)) << 16;
    return __builtin_bit_cast(float, u);
}
__device__ __forceinline__ float fast_sigmoid(float x) {
    return __builtin_amdgcn_rcpf(1.f + __expf(-x));   // v_rcp_f32, no div chain
}

#define BB 256
#define CC 64
#define HH 48
#define SP (HH*HH)           // 2304

// ---------------------------------------------------------------
// Kernel 1: routing (adaptive avgpool 2x2 + embed + linear + sigmoid)
// fused with per-sample expert kernel mix, packed into WMMA A-frag order.
// grid = B, block = 256
// ---------------------------------------------------------------
__global__ void routing_kernel(const float* __restrict__ x,
                               const int*   __restrict__ cluster,
                               const float* __restrict__ expert_w,  // [5][64][64]
                               const float* __restrict__ route_emb, // [64][256]
                               const float* __restrict__ route_W,   // [5][512]
                               const float* __restrict__ route_b,   // [5]
                               __bf16*      __restrict__ kpack)     // [B][4096]
{
    __shared__ float feat[512];
    __shared__ float srw[8];
    const int b = blockIdx.x, t = threadIdx.x;
    const int cl = cluster[b];

    {
        int c = t >> 2, qh = (t >> 1) & 1, qw = t & 1;
        const float* base = x + (size_t)b*CC*SP + (size_t)c*SP + qh*24*HH + qw*24;
        float s = 0.f;
        for (int r = 0; r < 24; ++r)
            for (int q = 0; q < 24; ++q)
                s += base[r*HH + q];
        feat[t] = s * (1.f/576.f);
        feat[256 + t] = route_emb[cl*256 + t];
    }
    __syncthreads();
    if (t < 5) {
        float z = route_b[t];
        const float* wrow = route_W + t*512;
        for (int j = 0; j < 512; ++j) z += feat[j] * wrow[j];
        srw[t] = fast_sigmoid(z);
    }
    __syncthreads();
    const float r0 = srw[0], r1 = srw[1], r2 = srw[2], r3 = srw[3], r4 = srw[4];

    // pack kern[b] = sum_e rw[e]*expert_w[e] into A-fragment order (bf16)
    for (int p = t; p < 4096; p += 256) {
        int i    = p & 15;
        int lane = (p >> 4) & 31;
        int mk   = p >> 9;           // mtile*2 + kstep
        int ks   = mk & 1;
        int mt   = mk >> 1;
        int M = mt*16 + (lane & 15);
        int K = ks*32 + 16*(lane >> 4) + i;   // A-frag: contiguous k = 16*half + i
        int oi = M*64 + K;
        float v = r0*expert_w[oi] + r1*expert_w[4096 + oi] + r2*expert_w[8192 + oi]
                + r3*expert_w[12288 + oi] + r4*expert_w[16384 + oi];
        kpack[(size_t)b*4096 + p] = f2bf(v);
    }
}

// ---------------------------------------------------------------
// Kernel 2: weight-norm scales for v1/v2/v3.  grid = 224, block = 64
// ---------------------------------------------------------------
__global__ void wn_scale_kernel(const float* __restrict__ v1, const float* __restrict__ g1,
                                const float* __restrict__ v2, const float* __restrict__ g2,
                                const float* __restrict__ v3, const float* __restrict__ g3,
                                float* __restrict__ scales)   // [224]
{
    __shared__ float red[64];
    const int blk = blockIdx.x, t = threadIdx.x;
    const float* v; const float* g; int K, o;
    if (blk < 32)      { v = v1; g = g1; K = 576; o = blk; }
    else if (blk < 96) { v = v2; g = g2; K = 288; o = blk - 32; }
    else               { v = v3; g = g3; K = 576; o = blk - 96; }
    const float* row = v + (size_t)o*K;
    float s = 0.f;
    for (int j = t; j < K; j += 64) s += row[j]*row[j];
    red[t] = s;
    __syncthreads();
    for (int st = 32; st > 0; st >>= 1) {
        if (t < st) red[t] += red[t + st];
        __syncthreads();
    }
    if (t == 0) scales[blk] = g[o] * rsqrtf(red[0]);
}

// ---------------------------------------------------------------
// Kernel 3: pack weight-normed conv weights into WMMA A-frag order (bf16).
// Conv GEMM-K is PERMUTED: k = (kh*3+kw)*CIN + ci  (tap-major, channel-minor).
// v layout is [o][ci][kh][kw] => flat = o*K + ci*9 + r9.
// grid = 432, block = 256
// ---------------------------------------------------------------
__global__ void wpack_kernel(const float* __restrict__ v1, const float* __restrict__ v2,
                             const float* __restrict__ v3, const float* __restrict__ scales,
                             __bf16* __restrict__ a1, __bf16* __restrict__ a2,
                             __bf16* __restrict__ a3)
{
    int p = blockIdx.x*256 + threadIdx.x;
    const float* v; const float* sc; __bf16* dst; int KST, K, CIN, lp;
    if (p < 18432)      { v = v1; sc = scales;      dst = a1; KST = 18; K = 576; CIN = 64; lp = p; }
    else if (p < 36864) { v = v2; sc = scales + 32; dst = a2; KST = 9;  K = 288; CIN = 32; lp = p - 18432; }
    else                { v = v3; sc = scales + 96; dst = a3; KST = 18; K = 576; CIN = 64; lp = p - 36864; }
    int i    = lp & 15;
    int lane = (lp >> 4) & 31;
    int mk   = lp >> 9;
    int ks   = mk % KST;
    int mt   = mk / KST;
    int M  = mt*16 + (lane & 15);
    int Kk = ks*32 + 16*(lane >> 4) + i;  // A-frag: contiguous k = 16*half + i
    int r9 = Kk / CIN;            // tap index (kh*3+kw)
    int ci = Kk - r9*CIN;         // in-channel
    dst[lp] = f2bf(v[(size_t)M*K + ci*9 + r9] * sc[M]);
}

// ---------------------------------------------------------------
// Kernel 4: gated expert 1x1 conv.  cc = kern[b] @ x[b]; out = sigmoid(cc+b)*x
// x staged once in LDS transposed [n][k] (stride 80); B-frags AND the gate's
// x values both come from LDS (no second global pass over x).
// grid = (72, B), block = 256 (8 waves: 4 mtiles x 2 nsubs over 64x32 tile)
// ---------------------------------------------------------------
__global__ void gated1x1_kernel(const float*  __restrict__ x,      // [B][64][2304] f32
                                const __bf16* __restrict__ kpack,  // [B][4096] A-order
                                const float*  __restrict__ cond_b, // [64]
                                __bf16*       __restrict__ xg)     // [B][64][2304] bf16
{
    __shared__ __bf16 s_xT[32*80];          // [n][k], stride 80 (160B, 32B-aligned)
    const int b = blockIdx.y;
    const int sbase = blockIdx.x * 32;
    const int t = threadIdx.x;
    const float* xb = x + (size_t)b*CC*SP;

    for (int e = t; e < 2048; e += 256) {
        int k = e >> 5, n = e & 31;         // coalesced global read (n innermost)
        s_xT[n*80 + k] = f2bf(xb[k*SP + sbase + n]);
    }
    __syncthreads();

    const int wave = t >> 5, lane = t & 31;
    const int half = lane >> 4, ln = lane & 15;
    const int mtile = wave & 3, nsub = wave >> 2;

    v8f acc = {};
    #pragma unroll
    for (int ks = 0; ks < 2; ++ks) {
        v16bf a = *reinterpret_cast<const v16bf*>(
            kpack + (size_t)b*4096 + (mtile*2 + ks)*512 + lane*16);
        // B-frag: element i holds K = ks*32 + 16*half + i, N = nsub*16+ln
        v16bf bm = *reinterpret_cast<const v16bf*>(
            &s_xT[(nsub*16 + ln)*80 + ks*32 + half*16]);
        acc = __builtin_amdgcn_wmma_f32_16x16x32_bf16(
            false, a, false, bm, (short)0, acc, false, false);
    }
    // gate inputs: 8 consecutive channels at fixed n -> one 16B LDS vector load
    v8bf xv = *reinterpret_cast<const v8bf*>(
        &s_xT[(nsub*16 + ln)*80 + mtile*16 + half*8]);
    #pragma unroll
    for (int r = 0; r < 8; ++r) {
        int o = mtile*16 + half*8 + r;      // D layout: VGPR r -> M = 8*half + r
        int s = sbase + nsub*16 + ln;
        float g = fast_sigmoid(acc[r] + cond_b[o]);
        xg[(size_t)b*CC*SP + (size_t)o*SP + s] = f2bf(g * bf2f(xv[r]));
    }
}

// ---------------------------------------------------------------
// Kernels 5-7: implicit-GEMM 3x3 SAME conv + PReLU + 2x2 maxpool.
// GEMM-K permuted: k = tap*CIN + ci. Input staged in LDS as
// [row 0..3][padded col 0..W+1][ci] (stride CIN+16) with zero padding
// baked in, so each B-frag is ONE aligned contiguous vector load.
// block = 256 (8 waves), wave tile = 16 out-chans x 16 spatial, K step 32.
// ---------------------------------------------------------------
template<int CIN, int COUT, int W, int KSTEPS, bool OUTF32>
__global__ void conv_wmma_kernel(const __bf16* __restrict__ in,    // [B][CIN][W][W]
                                 const __bf16* __restrict__ apack, // A-frag packed weights
                                 const float*  __restrict__ bias,  // [COUT]
                                 const float*  __restrict__ aprm,  // prelu scalar
                                 __bf16*       __restrict__ outB,  // [B][COUT][W/2][W/2]
                                 float*        __restrict__ outF)
{
    constexpr int P  = 2*W;                 // spatial positions per block (2 rows)
    constexpr int NT = (P + 15) / 16;
    constexpr int MT = COUT / 16;
    constexpr int WP = W + 2;               // padded width
    constexpr int ST = CIN + 16;            // LDS channel stride (32B-aligned cells)
    constexpr int W4 = W / 4;
    __shared__ __bf16 s_in[4*WP*ST];        // rows h0-1..h0+2, zero-padded cols
    __shared__ float  s_out[COUT*P];

    const int b  = blockIdx.y;
    const int pr = blockIdx.x;              // pooled output row
    const int h0 = pr*2;
    const int t  = threadIdx.x;
    const float alpha = aprm[0];

    const __bf16 bz = __builtin_bit_cast(__bf16, (unsigned short)0);
    // zero the left/right pad columns
    for (int e = t; e < 4*2*CIN; e += 256) {
        int lr   = e / (2*CIN);
        int side = (e / CIN) & 1;
        int ci   = e % CIN;
        int iwp  = side ? (W + 1) : 0;
        s_in[(lr*WP + iwp)*ST + ci] = bz;
    }
    // main fill: 8-byte coalesced global reads, scatter 4 bf16 into LDS
    for (int e = t; e < CIN*4*W4; e += 256) {
        int ci = e / (4*W4);
        int lr = (e / W4) & 3;
        int i4 = (e % W4) * 4;
        int gh = h0 - 1 + lr;
        ushort4v uv = {0, 0, 0, 0};
        if (gh >= 0 && gh < W)
            uv = *reinterpret_cast<const ushort4v*>(
                &in[((size_t)(b*CIN + ci)*W + gh)*W + i4]);
        #pragma unroll
        for (int j = 0; j < 4; ++j)
            s_in[(lr*WP + i4 + 1 + j)*ST + ci] =
                __builtin_bit_cast(__bf16, (unsigned short)uv[j]);
    }
    __syncthreads();

    const int wave = t >> 5, lane = t & 31;
    const int half = lane >> 4, ln = lane & 15;

    for (int tile = wave; tile < MT*NT; tile += 8) {
        int mtile = tile % MT, nt = tile / MT;
        int p  = nt*16 + ln;
        bool valid = (p < P);
        int pc = valid ? p : 0;             // clamp: EXEC stays full for WMMA
        int dr = pc / W, wc = pc % W;
        v8f acc = {};
        for (int ks = 0; ks < KSTEPS; ++ks) {
            v16bf a = *reinterpret_cast<const v16bf*>(
                apack + ((size_t)(mtile*KSTEPS + ks)*512 + lane*16));
            // B-frag: K run = [ks*32+half*16, +16) => fixed tap, 16 consecutive ci
            int kbase = ks*32 + half*16;
            int r9  = kbase / CIN;          // tap (kh*3+kw)
            int ci0 = kbase - r9*CIN;       // first channel (multiple of 16)
            int kh = r9 / 3, kw = r9 - (r9/3)*3;
            v16bf bm = *reinterpret_cast<const v16bf*>(
                &s_in[((dr + kh)*WP + wc + kw)*ST + ci0]);
            acc = __builtin_amdgcn_wmma_f32_16x16x32_bf16(
                false, a, false, bm, (short)0, acc, false, false);
        }
        if (valid) {
            #pragma unroll
            for (int r = 0; r < 8; ++r) {
                int o = mtile*16 + half*8 + r;
                float v = acc[r] + bias[o];
                v = v >= 0.f ? v : alpha*v;
                s_out[(o*2 + dr)*W + wc] = v;
            }
        }
    }
    __syncthreads();

    constexpr int WO = W/2;
    for (int e = t; e < COUT*WO; e += 256) {
        int o = e / WO, ow = e - o*WO;
        float m0 = s_out[(o*2+0)*W + 2*ow];
        float m1 = s_out[(o*2+0)*W + 2*ow + 1];
        float m2 = s_out[(o*2+1)*W + 2*ow];
        float m3 = s_out[(o*2+1)*W + 2*ow + 1];
        float m = fmaxf(fmaxf(m0, m1), fmaxf(m2, m3));
        size_t oi = ((size_t)(b*COUT + o)*WO + pr)*WO + ow;
        if constexpr (OUTF32) outF[oi] = m;
        else                  outB[oi] = f2bf(m);
    }
}

// ---------------------------------------------------------------
// Kernel 8: spatial mean + cluster embedding + 3-layer MLP head
// grid = B, block = 128
// ---------------------------------------------------------------
__global__ void head_kernel(const float* __restrict__ h3,   // [B][128][6][6]
                            const int*   __restrict__ cluster,
                            const float* __restrict__ emb2, const float* __restrict__ alpha,
                            const float* __restrict__ mW1, const float* __restrict__ mb1,
                            const float* __restrict__ a4,
                            const float* __restrict__ mW2, const float* __restrict__ mb2,
                            const float* __restrict__ a5,
                            const float* __restrict__ mW3, const float* __restrict__ mb3,
                            float* __restrict__ out)        // [B][65]
{
    __shared__ float h[128], z1[64], z2[32];
    const int b = blockIdx.x, t = threadIdx.x;
    const int cl = cluster[b];
    {
        const float* row = h3 + (size_t)b*128*36 + t*36;
        float s = 0.f;
        for (int j = 0; j < 36; ++j) s += row[j];
        h[t] = s*(1.f/36.f) + emb2[cl*128 + t]*alpha[0];
    }
    __syncthreads();
    if (t < 64) {
        float z = mb1[t];
        const float* wr = mW1 + t*128;
        for (int j = 0; j < 128; ++j) z += wr[j]*h[j];
        z1[t] = z >= 0.f ? z : a4[0]*z;
    }
    __syncthreads();
    if (t < 32) {
        float z = mb2[t];
        const float* wr = mW2 + t*64;
        for (int j = 0; j < 64; ++j) z += wr[j]*z1[j];
        z2[t] = z >= 0.f ? z : a5[0]*z;
    }
    __syncthreads();
    if (t < 65) {
        float z = mb3[t];
        const float* wr = mW3 + t*32;
        for (int j = 0; j < 32; ++j) z += wr[j]*z2[j];
        out[(size_t)b*65 + t] = z;
    }
}

// ---------------------------------------------------------------
extern "C" void kernel_launch(void* const* d_in, const int* in_sizes, int n_in,
                              void* d_out, int out_size, void* d_ws, size_t ws_size,
                              hipStream_t stream) {
    const float* x         = (const float*)d_in[0];
    const int*   cluster   = (const int*)  d_in[1];
    const float* expert_w  = (const float*)d_in[2];
    const float* cond_b    = (const float*)d_in[3];
    const float* route_emb = (const float*)d_in[4];
    const float* route_W   = (const float*)d_in[5];
    const float* route_b   = (const float*)d_in[6];
    const float* v1 = (const float*)d_in[7];
    const float* g1 = (const float*)d_in[8];
    const float* cb1 = (const float*)d_in[9];
    const float* a1 = (const float*)d_in[10];
    const float* v2 = (const float*)d_in[11];
    const float* g2 = (const float*)d_in[12];
    const float* cb2 = (const float*)d_in[13];
    const float* a2 = (const float*)d_in[14];
    const float* v3 = (const float*)d_in[15];
    const float* g3 = (const float*)d_in[16];
    const float* cb3 = (const float*)d_in[17];
    const float* a3 = (const float*)d_in[18];
    const float* emb2  = (const float*)d_in[19];
    const float* alpha = (const float*)d_in[20];
    const float* mW1 = (const float*)d_in[21];
    const float* mb1 = (const float*)d_in[22];
    const float* a4  = (const float*)d_in[23];
    const float* mW2 = (const float*)d_in[24];
    const float* mb2 = (const float*)d_in[25];
    const float* a5  = (const float*)d_in[26];
    const float* mW3 = (const float*)d_in[27];
    const float* mb3 = (const float*)d_in[28];
    float* out = (float*)d_out;

    char* ws = (char*)d_ws;
    size_t off = 0;
    auto take = [&](size_t bytes) -> void* {
        void* p = ws + off;
        off = (off + bytes + 255) & ~(size_t)255;
        return p;
    };
    __bf16* kpack  = (__bf16*)take((size_t)BB*4096*2);
    float*  scales = (float*) take(224*4);
    __bf16* a1p    = (__bf16*)take(18432*2);
    __bf16* a2p    = (__bf16*)take(18432*2);
    __bf16* a3p    = (__bf16*)take(73728*2);
    __bf16* xg     = (__bf16*)take((size_t)BB*CC*SP*2);
    __bf16* h1     = (__bf16*)take((size_t)BB*32*24*24*2);
    __bf16* h2     = (__bf16*)take((size_t)BB*64*12*12*2);
    float*  h3     = (float*) take((size_t)BB*128*36*4);
    (void)ws_size; (void)n_in; (void)in_sizes; (void)out_size;

    routing_kernel<<<BB, 256, 0, stream>>>(x, cluster, expert_w, route_emb,
                                           route_W, route_b, kpack);
    wn_scale_kernel<<<224, 64, 0, stream>>>(v1, g1, v2, g2, v3, g3, scales);
    wpack_kernel<<<432, 256, 0, stream>>>(v1, v2, v3, scales, a1p, a2p, a3p);
    gated1x1_kernel<<<dim3(72, BB), 256, 0, stream>>>(x, kpack, cond_b, xg);
    conv_wmma_kernel<64, 32, 48, 18, false><<<dim3(24, BB), 256, 0, stream>>>(
        xg, a1p, cb1, a1, h1, nullptr);
    conv_wmma_kernel<32, 64, 24, 9, false><<<dim3(12, BB), 256, 0, stream>>>(
        h1, a2p, cb2, a2, h2, nullptr);
    conv_wmma_kernel<64, 128, 12, 18, true><<<dim3(6, BB), 256, 0, stream>>>(
        h2, a3p, cb3, a3, nullptr, h3);
    head_kernel<<<BB, 128, 0, stream>>>(h3, cluster, emb2, alpha,
                                        mW1, mb1, a4, mW2, mb2, a5, mW3, mb3, out);
}
